// UNet_74792560493045
// MI455X (gfx1250) — matrix operand order repaired and checked by reading, not compiled, not run on hardware
//
#include <hip/hip_runtime.h>
#include <math.h>

// ---------------------------------------------------------------------------
// Types for CDNA5 WMMA
// ---------------------------------------------------------------------------
typedef __bf16 bf16_t;
typedef bf16_t v16bf __attribute__((ext_vector_type(16)));
typedef bf16_t v2bf  __attribute__((ext_vector_type(2)));
typedef float  v8f   __attribute__((ext_vector_type(8)));

// ---------------------------------------------------------------------------
// Permutation maps (replace the reference's dense permutation-matrix matmuls)
//   pM(i) = i ^ 2 ; pN(i) = identity pads, pair-swap inside
//   type 1 : gF = pM o pN   (r1/c1, r2/c2 stages; fwd gather == scatter map)
//   type 2 : pN             (N[2]/N[3] stages)
// ---------------------------------------------------------------------------
__device__ __forceinline__ int pN_map(int i, int n) {
    if (i < 2 || i >= n - 2) return i;
    return ((i - 2) ^ 2) + 2;
}
__device__ __forceinline__ int smap(int i, int n, int t) {
    if (t == 1) return pN_map(i, n) ^ 2;
    if (t == 2) return pN_map(i, n);
    return i;
}

// native bf16 pack (hardware convert path on gfx1250)
__device__ __forceinline__ unsigned pack_bf2(float lo, float hi) {
    union { v2bf h; unsigned u; } c;
    c.h.x = (bf16_t)lo;
    c.h.y = (bf16_t)hi;
    return c.u;
}

__device__ __forceinline__ float gelu_tanh(float x) {
    const float k = 0.7978845608028654f;  // sqrt(2/pi)
    float x3 = x * x * x;
    return 0.5f * x * (1.f + tanhf(k * (x + 0.044715f * x3)));
}

__device__ __forceinline__ float f4c(const float4& v, int j) {
    return j == 0 ? v.x : j == 1 ? v.y : j == 2 ? v.z : v.w;
}

// ---------------------------------------------------------------------------
// Direct 3x3 conv, pad=1, stride 1 or 2, optional relu, optional residual add.
// Halo handled branchlessly: clamped address + mask multiply.
// ---------------------------------------------------------------------------
__global__ __launch_bounds__(256) void conv3x3_kernel(
    const float* __restrict__ in, const float* __restrict__ w,
    const float* __restrict__ bias, float* __restrict__ out,
    const float* __restrict__ add, int Cin, int Cout, int Hin, int Win,
    int stride, int relu)
{
    int Hout = Hin / stride, Wout = Win / stride;
    int idx = blockIdx.x * 256 + threadIdx.x;
    int total = Cout * Hout * Wout;
    if (idx >= total) return;
    int wo = idx % Wout;
    int t = idx / Wout;
    int ho = t % Hout;
    int co = t / Hout;
    float acc = bias[co];
    for (int ci = 0; ci < Cin; ci++) {
        const float* ip = in + (size_t)ci * Hin * Win;
        const float* wp = w + ((size_t)co * Cin + ci) * 9;
        #pragma unroll
        for (int kh = 0; kh < 3; kh++) {
            int hi = ho * stride + kh - 1;
            int hic = min(max(hi, 0), Hin - 1);
            float mh = ((unsigned)hi < (unsigned)Hin) ? 1.f : 0.f;
            #pragma unroll
            for (int kw = 0; kw < 3; kw++) {
                int wi = wo * stride + kw - 1;
                int wic = min(max(wi, 0), Win - 1);
                float m = ((unsigned)wi < (unsigned)Win) ? mh : 0.f;
                acc += ip[hic * Win + wic] * wp[kh * 3 + kw] * m;
            }
        }
    }
    if (relu) acc = fmaxf(acc, 0.f);
    if (add) acc += add[idx];
    out[idx] = acc;
}

// ---------------------------------------------------------------------------
// Channel LayerNorm per pixel; optional fused forward permutation gather:
//   out[c, i, k] = LN(x)[c, smap(i), smap(k)]
// ---------------------------------------------------------------------------
__global__ __launch_bounds__(256) void ln_kernel(
    const float* __restrict__ x, const float* __restrict__ g,
    const float* __restrict__ b, float* __restrict__ out,
    int C, int H, int W, int shuf)
{
    int p = blockIdx.x * 256 + threadIdx.x;
    int HW = H * W;
    if (p >= HW) return;
    int i = p / W, k = p - i * W;
    int ps = smap(i, H, shuf) * W + smap(k, W, shuf);
    float s = 0.f, ss = 0.f;
    for (int c = 0; c < C; c++) {
        float v = x[(unsigned)(c * HW + ps)];
        s += v; ss += v * v;
    }
    float mu = s / (float)C;
    float var = ss / (float)C - mu * mu;
    float r = rsqrtf(var + 1e-6f);
    for (int c = 0; c < C; c++) {
        float v = (x[(unsigned)(c * HW + ps)] - mu) * r;
        out[(unsigned)(c * HW + p)] = v * g[c] + b[c];
    }
}

// ---------------------------------------------------------------------------
// WMMA GEMM pieces (software-pipelined, double-buffered LDS)
// ---------------------------------------------------------------------------
struct StageRegs {
    float a0, a1, a2, a3;   // A: 4 consecutive k of one m-row
    float4 b[4];            // B: 4 k-rows x 4 consecutive n
};

// Global -> registers. tail: zero-fill B rows k>=K; A only needs clamping
// (junk A rows >= M feed unstored D rows; junk A cols >= K hit zero B rows).
__device__ __forceinline__ void load_tile(
    StageRegs& R, int k0, bool tail,
    const float* __restrict__ arow, const float* __restrict__ Xn,
    int K, int N, int ak, int bkq)
{
    int ka0 = k0 + ak;
    if (!tail) {
        R.a0 = arow[ka0 + 0]; R.a1 = arow[ka0 + 1];
        R.a2 = arow[ka0 + 2]; R.a3 = arow[ka0 + 3];
        #pragma unroll
        for (int e = 0; e < 4; e++) {
            unsigned off = (unsigned)(k0 + bkq + e) * (unsigned)N;
            R.b[e] = *(const float4*)(Xn + off);
        }
    } else {
        R.a0 = arow[min(ka0 + 0, K - 1)];
        R.a1 = arow[min(ka0 + 1, K - 1)];
        R.a2 = arow[min(ka0 + 2, K - 1)];
        R.a3 = arow[min(ka0 + 3, K - 1)];
        #pragma unroll
        for (int e = 0; e < 4; e++) {
            int k = k0 + bkq + e;
            unsigned off = (unsigned)min(k, K - 1) * (unsigned)N;
            float4 v = *(const float4*)(Xn + off);
            if (k >= K) { v.x = 0.f; v.y = 0.f; v.z = 0.f; v.w = 0.f; }
            R.b[e] = v;
        }
    }
}

// Registers -> LDS (bf16 pack; A row-major [m][k], B k-major [n][k])
__device__ __forceinline__ void store_tile(
    const StageRegs& R, unsigned short* As, unsigned short* Bs,
    int am, int ak, int bn4, int bkq)
{
    uint2 sa;
    sa.x = pack_bf2(R.a0, R.a1);
    sa.y = pack_bf2(R.a2, R.a3);
    *(uint2*)(As + am * 40 + ak) = sa;
    #pragma unroll
    for (int j = 0; j < 4; j++) {
        uint2 sb;
        sb.x = pack_bf2(f4c(R.b[0], j), f4c(R.b[1], j));
        sb.y = pack_bf2(f4c(R.b[2], j), f4c(R.b[3], j));
        *(uint2*)(Bs + (bn4 + j) * 40 + bkq) = sb;
    }
}

// LDS fragments (2x ds_load_b128 each, contiguous k = 16h + s in A and B) + 2 WMMAs
__device__ __forceinline__ void mma_tile(
    const uint4* AsQ, const uint4* BsQ, int wm, int wn, int lane,
    v8f& acc0, v8f& acc1)
{
    union Frag { uint4 q[2]; v16bf v; } ua, ub0, ub1;
    int mr = wm * 16 + (lane & 15);
    int ka = (lane >> 4) * 16;                   // 0|16
    const uint4* ap = AsQ + mr * 5 + (ka >> 3);
    ua.q[0] = ap[0];                             // k: ka .. ka+7
    ua.q[1] = ap[1];                             // k: ka+8 .. ka+15

    int nc = lane & 15;
    int kb = (lane >> 4) * 16;                   // 0|16
    const uint4* bp0 = BsQ + (wn * 32 + nc) * 5 + (kb >> 3);
    ub0.q[0] = bp0[0]; ub0.q[1] = bp0[1];        // k: kb .. kb+15
    const uint4* bp1 = bp0 + 16 * 5;
    ub1.q[0] = bp1[0]; ub1.q[1] = bp1[1];

    acc0 = __builtin_amdgcn_wmma_f32_16x16x32_bf16(
        false, ua.v, false, ub0.v, (short)0, acc0, false, false);
    acc1 = __builtin_amdgcn_wmma_f32_16x16x32_bf16(
        false, ua.v, false, ub1.v, (short)0, acc1, false, false);
}

// ---------------------------------------------------------------------------
// WMMA GEMM:  D = Wm(MxK) @ X(KxN) [+ bias]
// 256 threads = 8 waves in 2(M) x 4(N); block tile 32(M) x 128(N); BK = 32.
// Double-buffered LDS, one barrier per K-step, prefetch overlaps WMMA.
// N is always a multiple of 128 at every call site.
// Epilogue modes:
//   0: out = val                    1: out = gelu(val)
//   2: resid[m, smap(h), smap(w)] += gelu(val)     (shuffle-back + residual)
//   3: resid += val                                (MLP residual)
//   4: pixel-shuffle r=2 scatter:  out[dst] = val + res_in[dst]
// ---------------------------------------------------------------------------
__global__ __launch_bounds__(256) void gemm_wmma_kernel(
    const float* __restrict__ Wm, const float* __restrict__ X,
    const float* __restrict__ bias, float* __restrict__ out,
    float* __restrict__ resid, const float* __restrict__ res_in,
    int M, int K, int N, int Wdim, int shuf, int mode)
{
    // per buffer: A = 160 uint4 (32 x 40 ushort), B = 640 uint4 (128 x 40)
    __shared__ uint4 LdsQ[2][800];

    const int tid  = threadIdx.x;
    const int lane = tid & 31;
    const int wid  = tid >> 5;
    const int wm   = wid >> 2;      // 0..1
    const int wn   = wid & 3;       // 0..3
    const int n0   = blockIdx.x * 128;
    const int m0   = blockIdx.y * 32;

    // staging assignments
    const int am  = tid >> 3;            // 0..31  A m-row
    const int ak  = (tid & 7) * 4;       // 0..28  A k-group
    const int bn4 = (tid & 31) * 4;      // 0..124 B n-group (float4)
    const int bkq = (tid >> 5) * 4;      // 0..28  B k-group of 4

    const int amc = min(m0 + am, M - 1); // clamp: junk rows are never stored
    const float* arow = Wm + (size_t)amc * K;
    const float* Xn   = X + n0 + bn4;

    v8f acc0 = {}, acc1 = {};

    const int T = (K + 31) >> 5;         // number of K tiles
    const int rem = K & 31;

    StageRegs R;
    load_tile(R, 0, (T == 1) && rem, arow, Xn, K, N, ak, bkq);
    store_tile(R, (unsigned short*)LdsQ[0], (unsigned short*)(LdsQ[0] + 160),
               am, ak, bn4, bkq);
    __syncthreads();

    for (int t = 0; t < T; t++) {
        int cb = t & 1;
        if (t + 1 < T)
            load_tile(R, (t + 1) * 32, (t + 1 == T - 1) && rem,
                      arow, Xn, K, N, ak, bkq);
        mma_tile(LdsQ[cb], LdsQ[cb] + 160, wm, wn, lane, acc0, acc1);
        if (t + 1 < T)
            store_tile(R, (unsigned short*)LdsQ[cb ^ 1],
                       (unsigned short*)(LdsQ[cb ^ 1] + 160), am, ak, bn4, bkq);
        __syncthreads();
    }

    // ---- epilogue (n always < N: N multiple of 128) ----
    int nc = lane & 15;
    int mb = (lane >> 4) * 8;
    #pragma unroll
    for (int half = 0; half < 2; half++) {
        v8f a = half ? acc1 : acc0;
        int n = n0 + wn * 32 + half * 16 + nc;
        #pragma unroll
        for (int e = 0; e < 8; e++) {
            int m = m0 + wm * 16 + mb + e;
            if (m >= M) continue;
            float val = a[e];
            if (bias) val += bias[m];
            if (mode == 0) {
                out[(unsigned)(m * N + n)] = val;
            } else if (mode == 1) {
                out[(unsigned)(m * N + n)] = gelu_tanh(val);
            } else if (mode == 2) {
                val = gelu_tanh(val);
                int h = n / Wdim, w = n - h * Wdim;
                int dh = smap(h, Wdim, shuf), dw = smap(w, Wdim, shuf);
                resid[(unsigned)(m * N + dh * Wdim + dw)] += val;
            } else if (mode == 3) {
                resid[(unsigned)(m * N + n)] += val;
            } else { // mode 4: pixel shuffle r=2 + skip add
                int cp = m >> 2, ii = (m >> 1) & 1, jj = m & 1;
                int h = n / Wdim, w = n - h * Wdim;
                int W2 = Wdim * 2;
                unsigned di = (unsigned)(cp * (4 * N) + (h * 2 + ii) * W2 + (w * 2 + jj));
                out[di] = val + res_in[di];
            }
        }
    }
}

// ---------------------------------------------------------------------------
// Host orchestration
// ---------------------------------------------------------------------------
struct BlockP {
    const float *g1, *b1, *pw, *pb, *g2, *b2, *w1, *bb1, *w2, *bb2;
};

extern "C" void kernel_launch(void* const* d_in, const int* in_sizes, int n_in,
                              void* d_out, int out_size, void* d_ws, size_t ws_size,
                              hipStream_t stream) {
    (void)out_size; (void)ws_size;
    static const int encN[4] = {2, 4, 12, 2};
    static const int decN[4] = {2, 12, 4, 2};

    const float* x = nullptr;
    const float *ciw = nullptr, *cib = nullptr, *cow = nullptr, *cob = nullptr;
    const float *dw[3], *db[3], *up[3];
    BlockP enc[4][12], dec[4][12];

    int cur = 0;
    auto nx = [&]() -> const float* {
        const float* p = (cur < n_in) ? (const float*)d_in[cur] : nullptr;
        cur++;
        return p;
    };

    bool insertion = (n_in > 0 && in_sizes[0] == 3 * 256 * 256);
    if (insertion) {
        // plain dict-insertion-order flatten: x first, then params
        x = nx();
        ciw = nx(); cib = nx(); cow = nx(); cob = nx();
        for (int i = 0; i < 3; i++) { dw[i] = nx(); db[i] = nx(); }
        up[0] = nx(); up[1] = nx(); up[2] = nx();
        for (int s = 0; s < 4; s++)
            for (int b = 0; b < encN[s]; b++) {
                BlockP& p = enc[s][b];
                p.g1 = nx(); p.b1 = nx(); p.pw = nx(); p.pb = nx();
                p.g2 = nx(); p.b2 = nx(); p.w1 = nx(); p.bb1 = nx();
                p.w2 = nx(); p.bb2 = nx();
            }
        for (int s = 0; s < 4; s++)
            for (int b = 0; b < decN[s]; b++) {
                BlockP& p = dec[s][b];
                p.g1 = nx(); p.b1 = nx(); p.pw = nx(); p.pb = nx();
                p.g2 = nx(); p.b2 = nx(); p.w1 = nx(); p.bb1 = nx();
                p.w2 = nx(); p.bb2 = nx();
            }
    } else {
        // jax.tree_leaves order: dict keys sorted; 'params' < 'x'
        // params: conv_in_b, conv_in_w, conv_out_b, conv_out_w, dec, down, enc, up
        // block keys sorted: b1,b2,bb1,bb2,g1,g2,pb,pw,w1,w2 ; down: b,w
        cib = nx(); ciw = nx(); cob = nx(); cow = nx();
        for (int s = 0; s < 4; s++)
            for (int b = 0; b < decN[s]; b++) {
                BlockP& p = dec[s][b];
                p.b1 = nx(); p.b2 = nx(); p.bb1 = nx(); p.bb2 = nx();
                p.g1 = nx(); p.g2 = nx(); p.pb = nx(); p.pw = nx();
                p.w1 = nx(); p.w2 = nx();
            }
        for (int i = 0; i < 3; i++) { db[i] = nx(); dw[i] = nx(); }
        for (int s = 0; s < 4; s++)
            for (int b = 0; b < encN[s]; b++) {
                BlockP& p = enc[s][b];
                p.b1 = nx(); p.b2 = nx(); p.bb1 = nx(); p.bb2 = nx();
                p.g1 = nx(); p.g2 = nx(); p.pb = nx(); p.pw = nx();
                p.w1 = nx(); p.w2 = nx();
            }
        up[0] = nx(); up[1] = nx(); up[2] = nx();
        x = nx();
    }

    // ---- workspace carve (floats). S1 = largest activation tensor ----
    const size_t S1 = (size_t)42 * 256 * 256;  // 2,752,512 floats
    float* ws   = (float*)d_ws;
    float* bufA = ws;
    float* bufB = ws + S1;
    float* tbuf = ws + 2 * S1;          // LN output           (<= S1)
    float* hbuf = ws + 3 * S1;          // MLP hidden 2C x HW  (<= 2*S1)
    float* res1 = ws + 5 * S1;          // 42 x 256^2
    float* res2 = ws + 6 * S1;          // 84 x 128^2  (S1/2)
    float* res3 = ws + 6 * S1 + S1 / 2; // 168 x 64^2  (S1/4)

    auto conv3 = [&](const float* in, const float* w, const float* b, float* out,
                     const float* add, int Cin, int Cout, int Hin, int stride, int relu) {
        int Hout = Hin / stride;
        int total = Cout * Hout * Hout;
        conv3x3_kernel<<<(total + 255) / 256, 256, 0, stream>>>(
            in, w, b, out, add, Cin, Cout, Hin, Hin, stride, relu);
    };
    auto gemm = [&](const float* Wm, const float* X, const float* bias, float* out,
                    float* resid, const float* resin,
                    int M, int K, int N, int Wdim, int shuf, int mode) {
        dim3 grid((N + 127) / 128, (M + 31) / 32);
        gemm_wmma_kernel<<<grid, 256, 0, stream>>>(
            Wm, X, bias, out, resid, resin, M, K, N, Wdim, shuf, mode);
    };
    auto ms_block = [&](float* act, const BlockP& p, int C, int H, int shuf) {
        int HW = H * H;
        // spatial-shuffle branch
        ln_kernel<<<(HW + 255) / 256, 256, 0, stream>>>(act, p.g1, p.b1, tbuf, C, H, H, shuf);
        gemm(p.pw, tbuf, p.pb, nullptr, act, nullptr, C, C, HW, H, shuf, 2);
        // channel MLP branch
        ln_kernel<<<(HW + 255) / 256, 256, 0, stream>>>(act, p.g2, p.b2, tbuf, C, H, H, 0);
        gemm(p.w1, tbuf, p.bb1, hbuf, nullptr, nullptr, 2 * C, C, HW, H, 0, 1);
        gemm(p.w2, hbuf, p.bb2, nullptr, act, nullptr, C, 2 * C, HW, H, 0, 3);
    };

    // ================= forward schedule =================
    conv3(x, ciw, cib, bufA, nullptr, 3, 42, 256, 1, 1);
    for (int b = 0; b < encN[0]; b++) ms_block(bufA, enc[0][b], 42, 256, 1);

    hipMemcpyAsync(res1, bufA, S1 * sizeof(float), hipMemcpyDeviceToDevice, stream);
    conv3(res1, dw[0], db[0], bufA, nullptr, 42, 84, 256, 2, 1);
    for (int b = 0; b < encN[1]; b++) ms_block(bufA, enc[1][b], 84, 128, 1);

    hipMemcpyAsync(res2, bufA, (size_t)84 * 128 * 128 * sizeof(float),
                   hipMemcpyDeviceToDevice, stream);
    conv3(res2, dw[1], db[1], bufA, nullptr, 84, 168, 128, 2, 1);
    for (int b = 0; b < encN[2]; b++) ms_block(bufA, enc[2][b], 168, 64, 2);

    hipMemcpyAsync(res3, bufA, (size_t)168 * 64 * 64 * sizeof(float),
                   hipMemcpyDeviceToDevice, stream);
    conv3(res3, dw[2], db[2], bufA, nullptr, 168, 336, 64, 2, 1);
    for (int b = 0; b < encN[3]; b++) ms_block(bufA, enc[3][b], 336, 32, 2);

    // decoder
    for (int b = 0; b < decN[0]; b++) ms_block(bufA, dec[0][b], 336, 32, 2);
    gemm(up[0], bufA, nullptr, bufB, nullptr, res3, 672, 336, 1024, 32, 0, 4);
    for (int b = 0; b < decN[1]; b++) ms_block(bufB, dec[1][b], 168, 64, 2);
    gemm(up[1], bufB, nullptr, bufA, nullptr, res2, 336, 168, 4096, 64, 0, 4);
    for (int b = 0; b < decN[2]; b++) ms_block(bufA, dec[2][b], 84, 128, 1);
    gemm(up[2], bufA, nullptr, bufB, nullptr, res1, 168, 84, 16384, 128, 0, 4);
    for (int b = 0; b < decN[3]; b++) ms_block(bufB, dec[3][b], 42, 256, 1);

    conv3(bufB, cow, cob, (float*)d_out, x, 42, 3, 256, 1, 0);
}
